// SelfAttention_49624052138298
// MI455X (gfx1250) — matrix operand order repaired and checked by reading, not compiled, not run on hardware
//
#include <hip/hip_runtime.h>

// ---------------- problem constants ----------------
#define EMBED   2048
#define NHEADS  32
#define NKV     8
#define HDIM    64
#define SEQ     2048
#define BATCH   2
#define NQKV    3072               // 2048 q + 512 k + 512 v
#define BTOK    (BATCH * SEQ)      // 4096 rows

// ---------------- vector types ----------------
typedef __attribute__((ext_vector_type(16))) __bf16       bf16x16;
typedef __attribute__((ext_vector_type(8)))  float        f32x8;
typedef __attribute__((ext_vector_type(4)))  unsigned int u32x4;
typedef __attribute__((ext_vector_type(4)))  unsigned int tdm_g0_t;
typedef __attribute__((ext_vector_type(8)))  int          tdm_g1_t;
typedef __attribute__((ext_vector_type(4)))  int          tdm_g2_t;
typedef __attribute__((ext_vector_type(8)))  int          tdm_g4_t;

union Frag { u32x4 u[2]; bf16x16 bf; };

__device__ __forceinline__ unsigned short f2bf(float f) {
  unsigned int u = __float_as_uint(f);
  u += 0x7FFFu + ((u >> 16) & 1u);           // round-to-nearest-even
  return (unsigned short)(u >> 16);
}

__device__ __forceinline__ f32x8 wmma_bf16(const Frag& a, const Frag& b, f32x8 c) {
  return __builtin_amdgcn_wmma_f32_16x16x32_bf16(
      false, a.bf, false, b.bf, (short)0, c, false, false);
}

// TDM: DMA a [128 rows x 32 cols] bf16 tile (row stride = k_total elems) from
// global into LDS with hardware padding 64B data + 16B pad per row, i.e. the
// LDS layout ushort[128][40].  Descriptor per CDNA5 ISA ch.8.
__device__ __forceinline__ void tdm_load_tile_128x32(
    const unsigned short* gtile, unsigned lds_byte_off,
    unsigned k_total, unsigned rows_total) {
  unsigned long long ga = (unsigned long long)(size_t)gtile;
  tdm_g0_t g0 = { 1u,                                   // count=1, user mode
                  lds_byte_off,                         // lds_addr
                  (unsigned)ga,                         // global_addr[31:0]
                  (unsigned)((ga >> 32) & 0x01FFFFFFu)  // global_addr[56:32]
                      | 0x80000000u };                  // type=2 ("image")
  tdm_g1_t g1 = { (int)((1u << 16)      // data_size = 2 bytes
                      | (1u << 20)      // pad_enable
                      | (3u << 22)      // pad_interval: 16 DW = 64B
                      | (3u << 25)),    // pad_amount:    4 DW = 16B
                  (int)(k_total << 16),                       // tensor_dim0 lo16
                  (int)((k_total >> 16) | (rows_total << 16)),// dim0 hi / dim1 lo
                  (int)((rows_total >> 16) | (32u << 16)),    // dim1 hi / tile_dim0
                  (int)128,                                   // tile_dim1 (|tile_dim2=0)
                  (int)k_total,                               // tensor_dim0_stride lo32
                  0, 0 };
  tdm_g2_t gz = { 0, 0, 0, 0 };
  tdm_g4_t gz8 = { 0, 0, 0, 0, 0, 0, 0, 0 };
  __builtin_amdgcn_tensor_load_to_lds(g0, g1, gz, gz, gz8, 0);
}

// ---------------- pack / convert kernels ----------------
__global__ void k_f32_to_bf16_vec(const float* __restrict__ in,
                                  unsigned short* __restrict__ out, int n4) {
  int i = blockIdx.x * blockDim.x + threadIdx.x;
  if (i >= n4) return;
  const float4 f = ((const float4*)in)[i];
  unsigned long long p = (unsigned long long)f2bf(f.x)
                       | ((unsigned long long)f2bf(f.y) << 16)
                       | ((unsigned long long)f2bf(f.z) << 32)
                       | ((unsigned long long)f2bf(f.w) << 48);
  ((unsigned long long*)out)[i] = p;
}

// WqkvT[n][k] = concat(Wq,Wk,Wv)[k][n]  (bf16, pre-transposed for WMMA B-frags)
__global__ void k_pack_wqkvT(const float* __restrict__ Wq, const float* __restrict__ Wk,
                             const float* __restrict__ Wv, unsigned short* __restrict__ WT) {
  int idx = blockIdx.x * blockDim.x + threadIdx.x;
  if (idx >= EMBED * NQKV) return;
  int k = idx / NQKV;
  int n = idx - k * NQKV;
  float w;
  if      (n < 2048) w = Wq[(size_t)k * 2048 + n];
  else if (n < 2560) w = Wk[(size_t)k * 512 + (n - 2048)];
  else               w = Wv[(size_t)k * 512 + (n - 2560)];
  WT[(size_t)n * EMBED + k] = f2bf(w);
}

__global__ void k_pack_woT(const float* __restrict__ Wo, unsigned short* __restrict__ WT) {
  int idx = blockIdx.x * blockDim.x + threadIdx.x;
  if (idx >= EMBED * EMBED) return;
  int k = idx / EMBED;
  int n = idx - k * EMBED;
  WT[(size_t)n * EMBED + k] = f2bf(Wo[(size_t)k * EMBED + n]);
}

__global__ void k_pack_bias(const float* __restrict__ bq, const float* __restrict__ bk,
                            const float* __restrict__ bv, float* __restrict__ bqkv) {
  int i = blockIdx.x * blockDim.x + threadIdx.x;
  if (i >= NQKV) return;
  bqkv[i] = (i < 2048) ? bq[i] : (i < 2560 ? bk[i - 2048] : bv[i - 2560]);
}

// ------- bf16 GEMM, 128x128 block, 8 waves x (64x32), TDM double-buffered ----
// C[M,N] (f32) = A[M,K](bf16) * BT[N,K](bf16)^T + bias[N]
__global__ __launch_bounds__(256) void k_gemm_bf16(
    const unsigned short* __restrict__ A,
    const unsigned short* __restrict__ BT,
    const float* __restrict__ bias,
    float* __restrict__ C,
    int M, int N, int K) {
  __shared__ __align__(16) unsigned short As[2][128][40]; // K-chunk 32, pad to 40
  __shared__ __align__(16) unsigned short Bs[2][128][40];

  const int m0   = blockIdx.x * 128;
  const int n0   = blockIdx.y * 128;
  const int tid  = threadIdx.x;
  const int lane = tid & 31;
  const int wid  = tid >> 5;
  const int wm   = wid >> 2;     // 0..1  (64-row strip)
  const int wn   = wid & 3;      // 0..3  (32-col strip)
  const int l15  = lane & 15;
  const int hi   = lane >> 4;

  const unsigned asOff = (unsigned)(size_t)(&As[0][0][0]);
  const unsigned bsOff = (unsigned)(size_t)(&Bs[0][0][0]);
  const unsigned bufBytes = 128u * 40u * 2u;              // 10240 B per buffer

  f32x8 acc[4][2];
#pragma unroll
  for (int i = 0; i < 4; i++)
#pragma unroll
    for (int j = 0; j < 2; j++)
      acc[i][j] = (f32x8){0.f, 0.f, 0.f, 0.f, 0.f, 0.f, 0.f, 0.f};

  // prologue: wave 0 issues the TDM loads for chunk 0 into buffer 0
  if (wid == 0) {
    tdm_load_tile_128x32(A  + (size_t)m0 * K, asOff, (unsigned)K, (unsigned)M);
    tdm_load_tile_128x32(BT + (size_t)n0 * K, bsOff, (unsigned)K, (unsigned)N);
  }

  for (int kc = 0; kc < K; kc += 32) {
    const int cur = (kc >> 5) & 1;
    if (wid == 0) {
      if (kc + 32 < K) {
        // issue next chunk into the other buffer, then wait for current chunk
        const unsigned nxt = cur ^ 1;
        tdm_load_tile_128x32(A  + (size_t)m0 * K + kc + 32,
                             asOff + nxt * bufBytes, (unsigned)K, (unsigned)M);
        tdm_load_tile_128x32(BT + (size_t)n0 * K + kc + 32,
                             bsOff + nxt * bufBytes, (unsigned)K, (unsigned)N);
        __builtin_amdgcn_s_wait_tensorcnt(2);  // in-order => current chunk done
      } else {
        __builtin_amdgcn_s_wait_tensorcnt(0);
      }
    }
    __syncthreads();                            // publish buffer `cur`

    Frag fb[2];
#pragma unroll
    for (int ns = 0; ns < 2; ns++) {
      const int r = wn * 32 + ns * 16 + l15;    // B col = r, K 0-7/16-23 vs 8-15/24-31
      const int c = hi * 8;
      fb[ns].u[0] = *(const u32x4*)&Bs[cur][r][c];
      fb[ns].u[1] = *(const u32x4*)&Bs[cur][r][c + 16];
    }
#pragma unroll
    for (int ms = 0; ms < 4; ms++) {
      Frag fa;
      const int r = wm * 64 + ms * 16 + l15;
      const int c = hi * 8;                     // A: K 0-7/16-23 vs 8-15/24-31
      fa.u[0] = *(const u32x4*)&As[cur][r][c];
      fa.u[1] = *(const u32x4*)&As[cur][r][c + 16];
#pragma unroll
      for (int ns = 0; ns < 2; ns++)
        acc[ms][ns] = wmma_bf16(fa, fb[ns], acc[ms][ns]);
    }
    __syncthreads();                            // all waves done with `cur`
  }

#pragma unroll
  for (int ns = 0; ns < 2; ns++) {
    const int col = n0 + wn * 32 + ns * 16 + l15;
    const float bc = bias[col];
#pragma unroll
    for (int ms = 0; ms < 4; ms++) {
      const int rbase = m0 + wm * 64 + ms * 16 + hi * 8;
#pragma unroll
      for (int v = 0; v < 8; v++)
        C[(size_t)(rbase + v) * N + col] = acc[ms][ns][v] + bc;
    }
  }
}

// ---------------- RoPE + repack qkv(f32) -> q/k/v (bf16) ----------------
__global__ void k_rope_pack(const float* __restrict__ qkv,
                            unsigned short* __restrict__ qo,
                            unsigned short* __restrict__ ko,
                            unsigned short* __restrict__ vo) {
  int idx = blockIdx.x * blockDim.x + threadIdx.x;
  if (idx >= BTOK * (NQKV / 2)) return;
  const int bt = idx / (NQKV / 2);
  const int c  = (idx - bt * (NQKV / 2)) * 2;
  const int b  = bt >> 11;
  const int t  = bt & (SEQ - 1);
  const float x0 = qkv[(size_t)bt * NQKV + c];
  const float x1 = qkv[(size_t)bt * NQKV + c + 1];
  if (c < 2560) {                         // q + k regions get RoPE
    const int d = c & 63;
    const float inv = __expf(-(float)d * (9.210340371976184f / 64.0f));
    float sn, cs;
    __sincosf((float)t * inv, &sn, &cs);
    const float y0 = x0 * cs - x1 * sn;
    const float y1 = x1 * cs + x0 * sn;
    if (c < 2048) {
      const int hh = c >> 6;
      const size_t o = (((size_t)b * NHEADS + hh) * SEQ + t) * HDIM + d;
      qo[o] = f2bf(y0); qo[o + 1] = f2bf(y1);
    } else {
      const int kh = (c - 2048) >> 6;
      const size_t o = (((size_t)b * NKV + kh) * SEQ + t) * HDIM + d;
      ko[o] = f2bf(y0); ko[o + 1] = f2bf(y1);
    }
  } else {
    const int vh = (c - 2560) >> 6;
    const int d  = c & 63;
    const size_t o = (((size_t)b * NKV + vh) * SEQ + t) * HDIM + d;
    vo[o] = f2bf(x0); vo[o + 1] = f2bf(x1);
  }
}

// ---------------- flash attention: 4 waves, 64 Q-rows/WG, K-chunks of 32 ----
__global__ __launch_bounds__(128) void k_flash_attn(
    const unsigned short* __restrict__ Q,   // [B,NH,T,D]
    const unsigned short* __restrict__ Kg,  // [B,NKV,T,D]
    const unsigned short* __restrict__ Vg,  // [B,NKV,T,D]
    unsigned short* __restrict__ ctx) {     // [B,T,NH*D]
  __shared__ __align__(16) unsigned short Ks[32][72];      // 32 k-rows x 64 d
  __shared__ __align__(16) unsigned short VT[64][40];      // transposed [d][k-row]
  __shared__ __align__(16) unsigned short Pb[4][16][40];   // per-wave P staging

  const int b   = blockIdx.z;
  const int h   = blockIdx.y;
  const int q0  = blockIdx.x * 64;
  const int kvh = h >> 2;                  // KV_REPEAT = 4
  const int tid  = threadIdx.x;
  const int lane = tid & 31;
  const int wid  = tid >> 5;
  const int l15  = lane & 15;
  const int hi   = lane >> 4;

  const unsigned short* qrow =
      Q + ((size_t)(b * NHEADS + h) * SEQ + q0 + wid * 16 + l15) * HDIM;
  Frag fq[2];
#pragma unroll
  for (int dc = 0; dc < 2; dc++) {
    const unsigned short* p = qrow + dc * 32 + hi * 8;
    fq[dc].u[0] = *(const u32x4*)p;
    fq[dc].u[1] = *(const u32x4*)(p + 16);
  }

  const unsigned short* kbase = Kg + (size_t)(b * NKV + kvh) * SEQ * HDIM;
  const unsigned short* vbase = Vg + (size_t)(b * NKV + kvh) * SEQ * HDIM;

  f32x8 o[4];
#pragma unroll
  for (int t = 0; t < 4; t++) o[t] = (f32x8){0.f,0.f,0.f,0.f,0.f,0.f,0.f,0.f};
  float m[8], l[8];
#pragma unroll
  for (int v = 0; v < 8; v++) { m[v] = -3.0e38f; l[v] = 0.f; }

  const int nchunk = q0 / 32 + 2;          // causal coverage of rows q0..q0+63
  const int lrow = tid >> 2;               // 0..31
  const int lcol = (tid & 3) * 16;         // 0,16,32,48

  for (int kc = 0; kc < nchunk; kc++) {
    const int kb = kc * 32;
    // --- cooperative K load + V transpose into LDS ---
    {
      const unsigned short* ks = kbase + (size_t)(kb + lrow) * HDIM + lcol;
      u32x4 ka = *(const u32x4*)ks;
      u32x4 kb2 = *(const u32x4*)(ks + 8);
      *(u32x4*)&Ks[lrow][lcol]     = ka;
      *(u32x4*)&Ks[lrow][lcol + 8] = kb2;

      const unsigned short* vs = vbase + (size_t)(kb + lrow) * HDIM + lcol;
      union { u32x4 u; unsigned short s[8]; } t0, t1;
      t0.u = *(const u32x4*)vs;
      t1.u = *(const u32x4*)(vs + 8);
#pragma unroll
      for (int i = 0; i < 8; i++) {
        VT[lcol + i][lrow]     = t0.s[i];
        VT[lcol + 8 + i][lrow] = t1.s[i];
      }
      if (kc + 1 < nchunk) {               // prefetch next chunk (global_prefetch_b8)
        __builtin_prefetch(kbase + (size_t)(kb + 32 + lrow) * HDIM + lcol, 0, 0);
        __builtin_prefetch(vbase + (size_t)(kb + 32 + lrow) * HDIM + lcol, 0, 0);
      }
    }
    __syncthreads();

    // --- S = Q K^T  (16x32 per wave, 4 WMMAs) ---
    f32x8 s0 = (f32x8){0.f,0.f,0.f,0.f,0.f,0.f,0.f,0.f};
    f32x8 s1 = s0;
#pragma unroll
    for (int dc = 0; dc < 2; dc++) {
      Frag fk0, fk1;
      const int c = dc * 32 + hi * 8;
      fk0.u[0] = *(const u32x4*)&Ks[l15][c];
      fk0.u[1] = *(const u32x4*)&Ks[l15][c + 16];
      fk1.u[0] = *(const u32x4*)&Ks[16 + l15][c];
      fk1.u[1] = *(const u32x4*)&Ks[16 + l15][c + 16];
      s0 = wmma_bf16(fq[dc], fk0, s0);
      s1 = wmma_bf16(fq[dc], fk1, s1);
    }

    // --- online softmax (f32) ---
#pragma unroll
    for (int v = 0; v < 8; v++) {
      const int r = q0 + wid * 16 + v + 8 * hi;
      float a0 = s0[v] * 0.125f;
      float a1 = s1[v] * 0.125f;
      if (kb + l15 > r)      a0 = -3.0e38f;   // causal mask
      if (kb + 16 + l15 > r) a1 = -3.0e38f;
      float mx = fmaxf(a0, a1);
#pragma unroll
      for (int off = 8; off; off >>= 1) mx = fmaxf(mx, __shfl_xor(mx, off, 32));
      const float mn   = fmaxf(m[v], mx);
      const float corr = __expf(m[v] - mn);
      const float p0 = __expf(a0 - mn);
      const float p1 = __expf(a1 - mn);
      float rs = p0 + p1;
#pragma unroll
      for (int off = 8; off; off >>= 1) rs += __shfl_xor(rs, off, 32);
      l[v] = l[v] * corr + rs;
      m[v] = mn;
#pragma unroll
      for (int t = 0; t < 4; t++) o[t][v] *= corr;
      Pb[wid][v + 8 * hi][l15]      = f2bf(p0);
      Pb[wid][v + 8 * hi][16 + l15] = f2bf(p1);
    }

    // --- O += P V  (4 WMMAs; A = P from LDS, B = V^T from LDS) ---
    Frag fpa;
    {
      const int c = hi * 8;
      fpa.u[0] = *(const u32x4*)&Pb[wid][l15][c];
      fpa.u[1] = *(const u32x4*)&Pb[wid][l15][c + 16];
    }
#pragma unroll
    for (int t = 0; t < 4; t++) {
      Frag fv;
      const int r = t * 16 + l15;
      const int c = hi * 8;
      fv.u[0] = *(const u32x4*)&VT[r][c];
      fv.u[1] = *(const u32x4*)&VT[r][c + 16];
      o[t] = wmma_bf16(fpa, fv, o[t]);
    }
    __syncthreads();
  }

  // --- normalize + write ctx[B,T,NH*D] (bf16) ---
#pragma unroll
  for (int v = 0; v < 8; v++) {
    const int r = q0 + wid * 16 + v + 8 * hi;
    const float inv = 1.0f / l[v];
    const size_t base = ((size_t)b * SEQ + r) * (NHEADS * HDIM) + h * HDIM;
#pragma unroll
    for (int t = 0; t < 4; t++)
      ctx[base + t * 16 + l15] = f2bf(o[t][v] * inv);
  }
}

// ---------------- launch ----------------
extern "C" void kernel_launch(void* const* d_in, const int* in_sizes, int n_in,
                              void* d_out, int out_size, void* d_ws, size_t ws_size,
                              hipStream_t stream) {
  (void)in_sizes; (void)n_in; (void)out_size; (void)ws_size;
  const float* x  = (const float*)d_in[0];
  const float* Wq = (const float*)d_in[1];
  const float* bq = (const float*)d_in[2];
  const float* Wk = (const float*)d_in[3];
  const float* bk = (const float*)d_in[4];
  const float* Wv = (const float*)d_in[5];
  const float* bv = (const float*)d_in[6];
  const float* Wo = (const float*)d_in[7];
  const float* bo = (const float*)d_in[8];
  float* out = (float*)d_out;
  char* ws = (char*)d_ws;

  unsigned short* xbf   = (unsigned short*)(ws + 0);           // 16.78 MB
  unsigned short* wqkvT = (unsigned short*)(ws + 16777216);    // 12.58 MB
  unsigned short* woT   = (unsigned short*)(ws + 29360128);    //  8.39 MB
  float*          biasq = (float*)         (ws + 37748736);    // 12 KB
  float*          qkv   = (float*)         (ws + 37761024);    // 50.33 MB
  unsigned short* qbf   = (unsigned short*)(ws + 88092672);    // 16.78 MB
  unsigned short* kbf   = (unsigned short*)(ws + 104869888);   //  4.19 MB
  unsigned short* vbf   = (unsigned short*)(ws + 109064192);   //  4.19 MB
  unsigned short* ctx   = (unsigned short*)(ws + 113258496);   // 16.78 MB

  k_f32_to_bf16_vec<<<(BTOK * EMBED / 4 + 255) / 256, 256, 0, stream>>>(
      x, xbf, BTOK * EMBED / 4);
  k_pack_wqkvT<<<(EMBED * NQKV + 255) / 256, 256, 0, stream>>>(Wq, Wk, Wv, wqkvT);
  k_pack_woT<<<(EMBED * EMBED + 255) / 256, 256, 0, stream>>>(Wo, woT);
  k_pack_bias<<<(NQKV + 255) / 256, 256, 0, stream>>>(bq, bk, bv, biasq);

  dim3 g1(BTOK / 128, NQKV / 128);
  k_gemm_bf16<<<g1, 256, 0, stream>>>(xbf, wqkvT, biasq, qkv, BTOK, NQKV, EMBED);

  k_rope_pack<<<(BTOK * (NQKV / 2) + 255) / 256, 256, 0, stream>>>(qkv, qbf, kbf, vbf);

  dim3 g2(SEQ / 64, NHEADS, BATCH);
  k_flash_attn<<<g2, 128, 0, stream>>>(qbf, kbf, vbf, ctx);

  dim3 g3(BTOK / 128, EMBED / 128);
  k_gemm_bf16<<<g3, 256, 0, stream>>>(ctx, woT, bo, out, BTOK, EMBED, EMBED);
}
